// CRF_12635793785084
// MI455X (gfx1250) — compile-verified, matrix operand order checked by reading, not run on hardware
//
#include <hip/hip_runtime.h>

#define B_SZ  256
#define T_LEN 2048
#define NT    48
#define NTOT  50   // NT + 2 (start, end)

typedef __attribute__((ext_vector_type(16))) _Float16 v16h;
typedef __attribute__((ext_vector_type(8)))  float    v8f;

// ---------------------------------------------------------------------------
// Kernel 1: per-batch real path score + sequence length.
// real[b] = em_score + trans[start, tag0] + pair_score + trans[last_tag, end]
// ---------------------------------------------------------------------------
__global__ __launch_bounds__(256) void crf_score_kernel(
    const float* __restrict__ inputs, const float* __restrict__ trans,
    const int* __restrict__ mask, const int* __restrict__ tags,
    float* __restrict__ realOut, int* __restrict__ lenOut)
{
  const int b   = blockIdx.x;
  const int tid = threadIdx.x;
  __shared__ float sEm[256];
  __shared__ float sPair[256];
  __shared__ int   sLen[256];

  const int*   tagRow = tags   + (size_t)b * T_LEN;
  const int*   mRow   = mask   + (size_t)b * T_LEN;
  const float* inRow  = inputs + (size_t)b * T_LEN * NT;

  float em = 0.f, pair = 0.f;
  int   lc = 0;
  for (int t = tid; t < T_LEN; t += 256) {
    int   mk = mRow[t];
    int   tg = tagRow[t];
    float fm = (float)mk;
    em += inRow[(size_t)t * NT + tg] * fm;
    lc += mk;
    if (t >= 1) {
      int tp = tagRow[t - 1];
      pair += trans[tp * NTOT + tg] * fm;
    }
  }
  sEm[tid] = em; sPair[tid] = pair; sLen[tid] = lc;
  __syncthreads();
  for (int s = 128; s > 0; s >>= 1) {
    if (tid < s) {
      sEm[tid]   += sEm[tid + s];
      sPair[tid] += sPair[tid + s];
      sLen[tid]  += sLen[tid + s];
    }
    __syncthreads();
  }
  if (tid == 0) {
    int   len     = sLen[0];
    int   lastTag = tagRow[len - 1];
    float first   = trans[NT * NTOT + tagRow[0]];
    realOut[b] = sEm[0] + first + sPair[0] + trans[lastTag * NTOT + (NT + 1)];
    lenOut[b]  = len;
  }
}

// Load 24 floats for this lane's (column, hi-half) slice of one timestep row.
__device__ __forceinline__ void load24(const float* __restrict__ p, float out[24]) {
  #pragma unroll
  for (int g = 0; g < 3; ++g) {
    float4 a = *(const float4*)(p + 16 * g);
    float4 c = *(const float4*)(p + 16 * g + 4);
    out[g * 8 + 0] = a.x; out[g * 8 + 1] = a.y; out[g * 8 + 2] = a.z; out[g * 8 + 3] = a.w;
    out[g * 8 + 4] = c.x; out[g * 8 + 5] = c.y; out[g * 8 + 6] = c.z; out[g * 8 + 7] = c.w;
  }
}

// ---------------------------------------------------------------------------
// Kernel 2: forward algorithm. One wave (32 lanes) owns a 16-batch tile and
// carries the transposed state G[j][m]: lane L holds column m = L&15 and rows
// j = 16g + r + 8*(L>>4) in register slot [g*8+r].  Per step:
//   cmax -> p = exp(G-cmax) packed lane-locally into B fragments ->
//   D = Ehat x p via 6x v_wmma_f32_16x16x32_f16 -> G = cmax + log(D) + feat_t.
// D-layout == next-step B-layout, so no shuffles/LDS are needed per step.
// ---------------------------------------------------------------------------
__global__ __launch_bounds__(32) void crf_forward_kernel(
    const float* __restrict__ inputs, const float* __restrict__ trans,
    const int* __restrict__ lenIn, float* __restrict__ totalOut)
{
  const int lane = threadIdx.x;
  const int col  = lane & 15;
  const int hi   = lane >> 4;
  const int b    = blockIdx.x * 16 + col;

  // --- constant A fragments: Ahat[j][i] = exp(trans[i][j]), padded i>=48 -> 0
  v16h A[3][2];
  #pragma unroll
  for (int g = 0; g < 3; ++g) {
    #pragma unroll
    for (int c = 0; c < 2; ++c) {
      #pragma unroll
      for (int h = 0; h < 16; ++h) {
        int kl = (h < 8) ? (h + 8 * hi) : (16 + (h - 8) + 8 * hi);
        int i  = 32 * c + kl;
        int j  = 16 * g + col;
        float v = (i < NT) ? __expf(trans[i * NTOT + j]) : 0.f;
        A[g][c][h] = (_Float16)v;
      }
    }
  }

  // --- per-lane start/end transition slices in the G register layout
  float ts[24], te[24];
  #pragma unroll
  for (int g = 0; g < 3; ++g) {
    #pragma unroll
    for (int r = 0; r < 8; ++r) {
      int j = 16 * g + r + 8 * hi;
      ts[g * 8 + r] = trans[NT * NTOT + j];          // trans[start, j]
      te[g * 8 + r] = trans[j * NTOT + (NT + 1)];    // trans[j, end]
    }
  }

  const int    mylen = lenIn[b];
  const float* fb    = inputs + (size_t)b * T_LEN * NT + 8 * hi;

  // --- init: fwd0 = feat(0) + trans[start,:] + log(NT)
  float G[24];
  {
    float f0[24];
    load24(fb, f0);
    const float LOG_NT = 3.8712010109078911f;  // log(48)
    #pragma unroll
    for (int k = 0; k < 24; ++k) G[k] = f0[k] + ts[k] + LOG_NT;
  }

  // --- software pipeline, depth 2
  float fcur[24], fn1[24];
  load24(fb + (size_t)1 * NT, fcur);
  load24(fb + (size_t)2 * NT, fn1);

  for (int t = 1; t < T_LEN; ++t) {
    // stage t+2 loads + prefetch hint further ahead
    float fn2[24];
    int t2 = (t + 2 < T_LEN) ? (t + 2) : (T_LEN - 1);
    int tp = (t + 8 < T_LEN) ? (t + 8) : (T_LEN - 1);
    __builtin_prefetch(fb + (size_t)tp * NT, 0, 1);
    load24(fb + (size_t)t2 * NT, fn2);

    // per-column max over all 48 rows (lane-local 24 + partner lane xor 16)
    float cm = G[0];
    #pragma unroll
    for (int k = 1; k < 24; ++k) cm = fmaxf(cm, G[k]);
    cm = fmaxf(cm, __shfl_xor(cm, 16, 32));

    // p = exp(G - cmax), lane-locally packed into the two B fragments
    _Float16 ph[24];
    #pragma unroll
    for (int k = 0; k < 24; ++k) ph[k] = (_Float16)__expf(G[k] - cm);
    v16h b0, b1;
    #pragma unroll
    for (int r = 0; r < 8; ++r) {
      b0[r]     = ph[r];            // tile0 rows -> K 0..7 (+8*hi)
      b0[r + 8] = ph[8 + r];        // tile1 rows -> K 16..23 (+8*hi)
      b1[r]     = ph[16 + r];       // tile2 rows -> K 32..39 (+8*hi)
      b1[r + 8] = (_Float16)0.f;    // pad K 48..63
    }

    // q^T = Ehat x p^T : 3 output tiles x 2 K-chunks
    v8f acc0 = {}, acc1 = {}, acc2 = {};
    acc0 = __builtin_amdgcn_wmma_f32_16x16x32_f16(false, A[0][0], false, b0, (short)0, acc0, false, false);
    acc0 = __builtin_amdgcn_wmma_f32_16x16x32_f16(false, A[0][1], false, b1, (short)0, acc0, false, false);
    acc1 = __builtin_amdgcn_wmma_f32_16x16x32_f16(false, A[1][0], false, b0, (short)0, acc1, false, false);
    acc1 = __builtin_amdgcn_wmma_f32_16x16x32_f16(false, A[1][1], false, b1, (short)0, acc1, false, false);
    acc2 = __builtin_amdgcn_wmma_f32_16x16x32_f16(false, A[2][0], false, b0, (short)0, acc2, false, false);
    acc2 = __builtin_amdgcn_wmma_f32_16x16x32_f16(false, A[2][1], false, b1, (short)0, acc2, false, false);

    float qa[24];
    #pragma unroll
    for (int r = 0; r < 8; ++r) {
      qa[r]      = acc0[r];
      qa[8 + r]  = acc1[r];
      qa[16 + r] = acc2[r];
    }

    // G_new = cmax + log(q) + feat_t, frozen past the sequence length
    const bool keep = (t < mylen);
    #pragma unroll
    for (int k = 0; k < 24; ++k) {
      float nv = cm + __logf(qa[k]) + fcur[k];
      G[k] = keep ? nv : G[k];
    }

    // rotate pipeline
    #pragma unroll
    for (int k = 0; k < 24; ++k) { fcur[k] = fn1[k]; fn1[k] = fn2[k]; }
  }

  // total[b] = logsumexp_j(G[j] + trans[j,end])
  float mx = -3.0e38f;
  #pragma unroll
  for (int k = 0; k < 24; ++k) mx = fmaxf(mx, G[k] + te[k]);
  mx = fmaxf(mx, __shfl_xor(mx, 16, 32));
  float s = 0.f;
  #pragma unroll
  for (int k = 0; k < 24; ++k) s += __expf(G[k] + te[k] - mx);
  s += __shfl_xor(s, 16, 32);
  if (hi == 0) totalOut[b] = mx + __logf(s);
}

// ---------------------------------------------------------------------------
// Kernel 3: out = sum_b(total[b] - real[b]) / sum_b(len[b])
// ---------------------------------------------------------------------------
__global__ __launch_bounds__(256) void crf_final_kernel(
    const float* __restrict__ total, const float* __restrict__ realS,
    const int* __restrict__ lens, float* __restrict__ out)
{
  const int tid = threadIdx.x;
  __shared__ float sd[256];
  __shared__ float sl[256];
  sd[tid] = total[tid] - realS[tid];
  sl[tid] = (float)lens[tid];
  __syncthreads();
  for (int s = 128; s > 0; s >>= 1) {
    if (tid < s) { sd[tid] += sd[tid + s]; sl[tid] += sl[tid + s]; }
    __syncthreads();
  }
  if (tid == 0) out[0] = sd[0] / sl[0];
}

extern "C" void kernel_launch(void* const* d_in, const int* in_sizes, int n_in,
                              void* d_out, int out_size, void* d_ws, size_t ws_size,
                              hipStream_t stream) {
  const float* inputs = (const float*)d_in[0];   // (B, T, 48) f32
  const float* trans  = (const float*)d_in[1];   // (50, 50)   f32
  const int*   mask   = (const int*)d_in[2];     // (B, T)     i32
  const int*   tags   = (const int*)d_in[3];     // (B, T)     i32
  float* out = (float*)d_out;

  float* ws    = (float*)d_ws;
  float* total = ws;                 // 256 floats
  float* realS = ws + 256;           // 256 floats
  int*   lens  = (int*)(ws + 512);   // 256 ints

  crf_score_kernel<<<B_SZ, 256, 0, stream>>>(inputs, trans, mask, tags, realS, lens);
  crf_forward_kernel<<<B_SZ / 16, 32, 0, stream>>>(inputs, trans, lens, total);
  crf_final_kernel<<<1, 256, 0, stream>>>(total, realS, lens, out);
}
